// Mixture_6519760355972
// MI455X (gfx1250) — compile-verified, matrix operand order checked by reading, not run on hardware
//
#include <hip/hip_runtime.h>

// ---------------------------------------------------------------------------
// Problem constants (B=8192, D=1024, K=8, H=2048)
// ---------------------------------------------------------------------------
#define B_N     8192
#define D_DIM   1024
#define K_EXP   8
#define H_DIM   2048

#define TM      64          // M tile (rows of gathered X per block)
#define TN      128         // N tile (columns of H per block)
#define KC      64          // K chunk staged in LDS per iteration
#define NPAN    (H_DIM / TN)     // 16 N panels
#define MCAP    (B_N + K_EXP * TM)   // 8704 padded-row capacity
#define MTILES  (MCAP / TM)          // 136

#define LDX     72          // padded LDS row stride (elems) for X tile (128B + 16B pad)
#define LDW     72          // padded LDS row stride (elems) for W tile

typedef __bf16 bf16_t;
typedef bf16_t v16bf __attribute__((ext_vector_type(16)));
typedef float  v8f   __attribute__((ext_vector_type(8)));

// TDM descriptor group types (amdgpu-toolchain / clang-23: 6-arg builtin)
typedef unsigned int tdm_v4u __attribute__((ext_vector_type(4)));
typedef int          tdm_v8i __attribute__((ext_vector_type(8)));
typedef int          tdm_v4i __attribute__((ext_vector_type(4)));

#if defined(__has_builtin)
#if __has_builtin(__builtin_amdgcn_tensor_load_to_lds) && __has_builtin(__builtin_amdgcn_s_wait_tensorcnt)
#define USE_TDM 1
#endif
#endif
#ifndef USE_TDM
#define USE_TDM 0
#endif

union FragU { uint4 q[2]; v16bf v; };

__device__ __forceinline__ unsigned short f2bf(float f) {
    unsigned int u = __float_as_uint(f);
    unsigned int r = u + 0x7FFFu + ((u >> 16) & 1u);   // round-to-nearest-even
    return (unsigned short)(r >> 16);
}

// ---------------------------------------------------------------------------
// Workspace layout (bytes from d_ws base, all 64B aligned)
// ---------------------------------------------------------------------------
#define OFF_CLUSTER 0u          // int[8192]
#define OFF_RANK    32768u      // int[8192]
#define OFF_COUNTS  65536u      // int[8]
#define OFF_POFF    65600u      // int[9]
#define OFF_CNORM   65664u      // float[8]
#define OFF_ORDER   65792u      // int[8704]
#define OFF_PARTIAL 100864u     // float[16 * 8704]
#define OFF_XG      658176u     // ushort[8704 * 1024] (bf16 bits)

// ---------------------------------------------------------------------------
// 1) init: zero counts, set order[] = -1
// ---------------------------------------------------------------------------
__global__ __launch_bounds__(256) void init_kernel(int* order, int* counts) {
    int gid = blockIdx.x * 256 + threadIdx.x;
    if (gid < MCAP) order[gid] = -1;
    if (gid < K_EXP) counts[gid] = 0;
}

// ---------------------------------------------------------------------------
// 2) ||c_k||^2, one block per centroid
// ---------------------------------------------------------------------------
__global__ __launch_bounds__(256) void cnorm_kernel(const float* __restrict__ cent,
                                                    float* __restrict__ cnorm) {
    __shared__ float red[256];
    int k = blockIdx.x;
    float s = 0.f;
    for (int d = threadIdx.x; d < D_DIM; d += 256) {
        float c = cent[k * D_DIM + d];
        s += c * c;
    }
    red[threadIdx.x] = s;
    __syncthreads();
    for (int st = 128; st > 0; st >>= 1) {
        if (threadIdx.x < (unsigned)st) red[threadIdx.x] += red[threadIdx.x + st];
        __syncthreads();
    }
    if (threadIdx.x == 0) cnorm[k] = red[0];
}

// ---------------------------------------------------------------------------
// 3) routing: one wave (32 lanes) per sample; argmin_k (||c||^2 - 2 x.c)
// ---------------------------------------------------------------------------
__global__ __launch_bounds__(256) void route_kernel(const float* __restrict__ x,
                                                    const float* __restrict__ cent,
                                                    const float* __restrict__ cnorm,
                                                    int* __restrict__ cluster,
                                                    int* __restrict__ rankArr,
                                                    int* __restrict__ counts) {
    int w    = threadIdx.x >> 5;
    int lane = threadIdx.x & 31;
    int b    = blockIdx.x * 8 + w;

    float acc[K_EXP];
#pragma unroll
    for (int k = 0; k < K_EXP; ++k) acc[k] = 0.f;

    const float* xr = x + (size_t)b * D_DIM;
    for (int i = 0; i < D_DIM / 32; ++i) {
        int d = lane + 32 * i;
        float xv = xr[d];
#pragma unroll
        for (int k = 0; k < K_EXP; ++k)
            acc[k] = fmaf(xv, cent[k * D_DIM + d], acc[k]);
    }
#pragma unroll
    for (int k = 0; k < K_EXP; ++k) {
        float s = acc[k];
        s += __shfl_xor(s, 16, 32);
        s += __shfl_xor(s,  8, 32);
        s += __shfl_xor(s,  4, 32);
        s += __shfl_xor(s,  2, 32);
        s += __shfl_xor(s,  1, 32);
        acc[k] = s;
    }
    if (lane == 0) {
        float best = cnorm[0] - 2.f * acc[0];
        int bk = 0;
#pragma unroll
        for (int k = 1; k < K_EXP; ++k) {
            float sc = cnorm[k] - 2.f * acc[k];
            if (sc < best) { best = sc; bk = k; }
        }
        cluster[b] = bk;
        rankArr[b] = atomicAdd(&counts[bk], 1);
    }
}

// ---------------------------------------------------------------------------
// 4) tiny scan: padded per-expert offsets (multiples of TM)
// ---------------------------------------------------------------------------
__global__ void scan_kernel(const int* __restrict__ counts, int* __restrict__ poff) {
    if (threadIdx.x == 0 && blockIdx.x == 0) {
        int acc = 0;
        poff[0] = 0;
        for (int k = 0; k < K_EXP; ++k) {
            acc += (counts[k] + TM - 1) & ~(TM - 1);
            poff[k + 1] = acc;
        }
    }
}

// ---------------------------------------------------------------------------
// 5) gather + f32->bf16 convert into expert-sorted order; record row->sample
// ---------------------------------------------------------------------------
__global__ __launch_bounds__(256) void gather_kernel(const float* __restrict__ x,
                                                     const int* __restrict__ cluster,
                                                     const int* __restrict__ rankArr,
                                                     const int* __restrict__ poff,
                                                     int* __restrict__ order,
                                                     unsigned short* __restrict__ Xg) {
    int w    = threadIdx.x >> 5;
    int lane = threadIdx.x & 31;
    int b    = blockIdx.x * 8 + w;
    int k    = cluster[b];
    int pos  = poff[k] + rankArr[b];
    if (lane == 0) order[pos] = b;
    const float* xr = x + (size_t)b * D_DIM;
    unsigned short* xo = Xg + (size_t)pos * D_DIM;
    for (int i = 0; i < D_DIM / 32; ++i) {
        int c = lane + 32 * i;
        xo[c] = f2bf(xr[c]);
    }
}

// ---------------------------------------------------------------------------
// 6) fused MoE GEMM:  partial[np][row] = sum_{h in panel} tanh(x.W1 + b1) * W2
//    block tile M=64 x N=128, 8 waves in 2x4 (wm x wn) arrangement,
//    each wave: 2x2 wmma accumulators (f32 16x16, K=32 bf16 steps).
//    X tile staged via Tensor Data Mover (pad_enable reproduces LDX stride).
// ---------------------------------------------------------------------------
__global__ __launch_bounds__(256) void moe_gemm_kernel(const unsigned short* __restrict__ Xg,
                                                       const float* __restrict__ W1,
                                                       const float* __restrict__ b1,
                                                       const float* __restrict__ W2,
                                                       const int* __restrict__ poff,
                                                       float* __restrict__ partial) {
    __shared__ __align__(16) unsigned short Xs[TM][LDX];   // [row][k]   (bf16 bits)
    __shared__ __align__(16) unsigned short Ws[TN][LDW];   // [h][k] transposed (bf16 bits)
    __shared__ float llbuf[4][TM];                         // per wave-column row sums

    const int mt  = blockIdx.x;
    const int np  = blockIdx.y;
    const int m0  = mt * TM;
    const int tid  = threadIdx.x;
    const int lane = tid & 31;
    const int w    = tid >> 5;
    const int wm   = w & 1;     // 0..1 -> 32-row half
    const int wn   = w >> 1;    // 0..3 -> 32-col quarter

    if (m0 >= poff[K_EXP]) return;   // beyond padded rows: uniform exit

    int k = 0;
#pragma unroll
    for (int e = 0; e < K_EXP; ++e)
        if (m0 >= poff[e] && m0 < poff[e + 1]) k = e;

    const float* W1k = W1 + (size_t)k * D_DIM * H_DIM;
    const float* b1k = b1 + (size_t)k * H_DIM;
    const float* W2k = W2 + (size_t)k * H_DIM;
    const int n0 = np * TN;

#if USE_TDM
    const unsigned xs_lds = (unsigned)(unsigned long long)(const void*)&Xs[0][0];
#endif

    v8f acc[2][2] = {};

    for (int kb = 0; kb < D_DIM; kb += KC) {
        __syncthreads();
#if USE_TDM
        // --- stage X tile via Tensor Data Mover (wave 0 issues the DMA) --
        if (w == 0) {
            unsigned long long ga =
                (unsigned long long)(const void*)&Xg[(size_t)m0 * D_DIM + kb];
            tdm_v4u g0;
            g0.x = 1u;                                    // count=1 (valid), user mode
            g0.y = xs_lds;                                // lds_addr
            g0.z = (unsigned)(ga & 0xFFFFFFFFu);          // global_addr[31:0]
            g0.w = (unsigned)((ga >> 32) & 0x01FFFFFFu)   // global_addr[56:32]
                   | (2u << 30);                          // type=2 ("image")
            tdm_v8i g1;
            g1[0] = (int)((1u << 16)      // data_size = 2B (bf16)
                        | (1u << 20)      // pad_enable
                        | (4u << 22)      // pad_interval: 32 DWORDs (one 128B row)
                        | (3u << 25));    // pad_amount: 4 DWORDs (16B -> LDX=72)
            g1[1] = (int)((unsigned)KC << 16);   // tensor_dim0 = 64 elems
            g1[2] = (int)((unsigned)TM << 16);   // tensor_dim1 = 64 rows
            g1[3] = (int)((unsigned)KC << 16);   // tile_dim0 = 64 elems/row
            g1[4] = (int)(unsigned)TM;           // tile_dim1 = 64 rows, tile_dim2 = 0
            g1[5] = (int)(unsigned)D_DIM;        // tensor_dim0_stride = 1024 elems
            g1[6] = 0;
            g1[7] = 0;
            tdm_v4i z4 = {0, 0, 0, 0};
            tdm_v8i z8 = {0, 0, 0, 0, 0, 0, 0, 0};
            __builtin_amdgcn_tensor_load_to_lds(g0, g1, z4, z4, z8, 0);
        }
#else
        // --- stage X tile: 64x64 bf16, 8B per op -------------------------
#pragma unroll
        for (int i = 0; i < 4; ++i) {
            int idx = i * 256 + tid;        // 0..1023 quads
            int row = idx >> 4;
            int c4  = (idx & 15) << 2;
            uint2 v = *(const uint2*)&Xg[(size_t)(m0 + row) * D_DIM + kb + c4];
            *(uint2*)&Xs[row][c4] = v;
        }
#endif
        // --- stage W1 tile transposed + f32->bf16: 64x128 ----------------
#pragma unroll
        for (int i = 0; i < 8; ++i) {
            int idx = i * 256 + tid;        // 0..2047 float4 groups
            int d   = idx >> 5;             // 0..63
            int h4  = (idx & 31) << 2;      // 0..124
            float4 f = *(const float4*)&W1k[(size_t)(kb + d) * H_DIM + n0 + h4];
            Ws[h4 + 0][d] = f2bf(f.x);
            Ws[h4 + 1][d] = f2bf(f.y);
            Ws[h4 + 2][d] = f2bf(f.z);
            Ws[h4 + 3][d] = f2bf(f.w);
            if (kb + KC < D_DIM)            // prefetch next K-chunk into L2/L0
                __builtin_prefetch(&W1k[(size_t)(kb + KC + d) * H_DIM + n0 + h4], 0, 3);
        }
#if USE_TDM
        if (w == 0) __builtin_amdgcn_s_wait_tensorcnt(0);
#endif
        __syncthreads();
        // --- compute: 2 K-steps of 32 ------------------------------------
        const int koff = (lane >> 4) << 3;   // 0 or 8 per ISA A/B layout
#pragma unroll
        for (int ks = 0; ks < KC; ks += 32) {
            const int c0 = ks + koff;
            FragU a[2], bm[2];
#pragma unroll
            for (int mi = 0; mi < 2; ++mi) {
                int row = wm * 32 + mi * 16 + (lane & 15);
                a[mi].q[0] = *(const uint4*)&Xs[row][c0];
                a[mi].q[1] = *(const uint4*)&Xs[row][c0 + 16];
            }
#pragma unroll
            for (int ni = 0; ni < 2; ++ni) {
                int col = wn * 32 + ni * 16 + (lane & 15);
                bm[ni].q[0] = *(const uint4*)&Ws[col][c0];
                bm[ni].q[1] = *(const uint4*)&Ws[col][c0 + 16];
            }
#pragma unroll
            for (int mi = 0; mi < 2; ++mi)
#pragma unroll
                for (int ni = 0; ni < 2; ++ni)
                    acc[mi][ni] = __builtin_amdgcn_wmma_f32_16x16x32_bf16(
                        false, a[mi].v, false, bm[ni].v,
                        (short)0, acc[mi][ni], false, false);
        }
    }

    // --- fused epilogue: tanh(acc + b1) * W2, reduce over panel columns ---
    float psum[2][8];
#pragma unroll
    for (int mi = 0; mi < 2; ++mi)
#pragma unroll
        for (int j = 0; j < 8; ++j) psum[mi][j] = 0.f;

#pragma unroll
    for (int ni = 0; ni < 2; ++ni) {
        int hl = wn * 32 + ni * 16 + (lane & 15);
        float b1v = b1k[n0 + hl];
        float w2v = W2k[n0 + hl];
#pragma unroll
        for (int mi = 0; mi < 2; ++mi)
#pragma unroll
            for (int j = 0; j < 8; ++j) {
                float hv = tanhf(acc[mi][ni][j] + b1v);
                psum[mi][j] = fmaf(hv, w2v, psum[mi][j]);
            }
    }
    // reduce within each 16-lane half (C layout: lanes 0-15 rows j, 16-31 rows j+8)
#pragma unroll
    for (int mi = 0; mi < 2; ++mi)
#pragma unroll
        for (int j = 0; j < 8; ++j) {
            float s = psum[mi][j];
            s += __shfl_xor(s, 1, 32);
            s += __shfl_xor(s, 2, 32);
            s += __shfl_xor(s, 4, 32);
            s += __shfl_xor(s, 8, 32);
            if ((lane & 15) == 0) {
                int row = wm * 32 + mi * 16 + j + ((lane >> 4) << 3);
                llbuf[wn][row] = s;
            }
        }
    __syncthreads();
    if (tid < TM) {
        float s = llbuf[0][tid] + llbuf[1][tid] + llbuf[2][tid] + llbuf[3][tid];
        partial[(size_t)np * MCAP + m0 + tid] = s;
    }
}

// ---------------------------------------------------------------------------
// 7) final reduce over N panels + b2, scatter to original sample order
// ---------------------------------------------------------------------------
__global__ __launch_bounds__(256) void reduce_kernel(const int* __restrict__ order,
                                                     const int* __restrict__ cluster,
                                                     const float* __restrict__ b2,
                                                     const float* __restrict__ partial,
                                                     float* __restrict__ out) {
    int r = blockIdx.x * 256 + threadIdx.x;
    int b = order[r];
    if (b < 0) return;
    float s = b2[cluster[b]];
#pragma unroll
    for (int np = 0; np < NPAN; ++np)
        s += partial[(size_t)np * MCAP + r];
    out[b] = s;
}

// ---------------------------------------------------------------------------
// launcher
// ---------------------------------------------------------------------------
extern "C" void kernel_launch(void* const* d_in, const int* in_sizes, int n_in,
                              void* d_out, int out_size, void* d_ws, size_t ws_size,
                              hipStream_t stream) {
    const float* x    = (const float*)d_in[0];   // [B, D]
    const float* cent = (const float*)d_in[1];   // [K, D]
    const float* W1   = (const float*)d_in[2];   // [K, D, H]
    const float* b1   = (const float*)d_in[3];   // [K, H]
    const float* W2   = (const float*)d_in[4];   // [K, H]
    const float* b2   = (const float*)d_in[5];   // [K]
    float* out = (float*)d_out;

    char* ws = (char*)d_ws;
    int*            cluster = (int*)(ws + OFF_CLUSTER);
    int*            rankArr = (int*)(ws + OFF_RANK);
    int*            counts  = (int*)(ws + OFF_COUNTS);
    int*            poff    = (int*)(ws + OFF_POFF);
    float*          cnorm   = (float*)(ws + OFF_CNORM);
    int*            order   = (int*)(ws + OFF_ORDER);
    float*          partial = (float*)(ws + OFF_PARTIAL);
    unsigned short* Xg      = (unsigned short*)(ws + OFF_XG);

    init_kernel<<<dim3(MCAP / 256), dim3(256), 0, stream>>>(order, counts);
    cnorm_kernel<<<dim3(K_EXP), dim3(256), 0, stream>>>(cent, cnorm);
    route_kernel<<<dim3(B_N / 8), dim3(256), 0, stream>>>(x, cent, cnorm,
                                                          cluster, rankArr, counts);
    scan_kernel<<<dim3(1), dim3(32), 0, stream>>>(counts, poff);
    gather_kernel<<<dim3(B_N / 8), dim3(256), 0, stream>>>(x, cluster, rankArr,
                                                           poff, order, Xg);
    moe_gemm_kernel<<<dim3(MTILES, NPAN), dim3(256), 0, stream>>>(Xg, W1, b1, W2,
                                                                  poff, partial);
    reduce_kernel<<<dim3(MCAP / 256), dim3(256), 0, stream>>>(order, cluster, b2,
                                                              partial, out);
}
